// ViT_UNet_52166672777659
// MI455X (gfx1250) — matrix operand order repaired and checked by reading, not compiled, not run on hardware
//
#include <hip/hip_runtime.h>
#include <hip/hip_bf16.h>
#include <math.h>

typedef __attribute__((ext_vector_type(16))) __bf16 bf16x16;
typedef __attribute__((ext_vector_type(8)))  float  f32x8;
typedef __attribute__((ext_vector_type(4)))  float  f32x4;
typedef __attribute__((ext_vector_type(4)))  unsigned int   u32x4;
typedef __attribute__((ext_vector_type(2)))  unsigned int   u32x2;
typedef __attribute__((ext_vector_type(4)))  unsigned short u16x4;
typedef __attribute__((ext_vector_type(8)))  int            i32x8;
typedef __attribute__((ext_vector_type(4)))  int            i32x4;
typedef unsigned short ushort_t;

#define DEVFN __device__ __forceinline__

DEVFN ushort_t f2bf(float x) {
  union { float f; unsigned int u; } c; c.f = x;
  unsigned int r = c.u + 0x7fffu + ((c.u >> 16) & 1u);  // round-to-nearest-even
  return (ushort_t)(r >> 16);
}

DEVFN float wave_sum(float v) {
#pragma unroll
  for (int off = 16; off > 0; off >>= 1) v += __shfl_xor(v, off, 32);
  return v;
}
DEVFN float wave_max(float v) {
#pragma unroll
  for (int off = 16; off > 0; off >>= 1) v = fmaxf(v, __shfl_xor(v, off, 32));
  return v;
}

// ---------------------------------------------------------------------------
// TDM: async-load a 2D tile [32 K-elems contiguous] x [128 rows, stride lda]
// of bf16 from global into LDS with 80-byte row pitch (pad 4 DW per 16 DW).
// Descriptor packing per CDNA5 ISA 8.3/8.4 (D# group0/group1).
// clang-23 toolchain: 6-arg builtin (g0, g1, g2, g3, extra, cpol).
// ---------------------------------------------------------------------------
DEVFN void tdm_load_2d(unsigned lds_off, const ushort_t* gptr, int lda_elems) {
  unsigned long long ga = (unsigned long long)(size_t)gptr;
  u32x4 g0;
  g0[0] = 1u;                                    // count=1, user descriptor
  g0[1] = lds_off;                               // lds_addr [63:32]
  g0[2] = (unsigned)(ga & 0xffffffffu);          // global_addr [95:64]
  g0[3] = (unsigned)((ga >> 32) & 0x01ffffffu)   // global_addr [120:96]
          | (2u << 30);                          // type=2 ("image")
  i32x8 g1;
  g1[0] = (1 << 16)    // data_size = 2 bytes
        | (1 << 20)    // pad_enable
        | (3 << 22)    // pad_interval: 16 DWORDs (64B = one 32-elem row)
        | (3 << 25);   // pad_amount: 4 DWORDs (16B) -> 80B pitch
  g1[1] = (32 << 16);        // tensor_dim0 = 32  (bits 79:48, low half)
  g1[2] = (128 << 16);       // tensor_dim0 hi=0 | tensor_dim1 = 128
  g1[3] = (32 << 16);        // tensor_dim1 hi=0 | tile_dim0 = 32
  g1[4] = 128;               // tile_dim1 = 128 | tile_dim2 = 0
  g1[5] = lda_elems;         // tensor_dim0_stride [31:0] (data_size units)
  g1[6] = 0;                 // stride0 hi | stride1 lo
  g1[7] = 0;
  i32x4 z4; z4[0] = 0; z4[1] = 0; z4[2] = 0; z4[3] = 0;
  i32x8 z8;
#pragma unroll
  for (int i = 0; i < 8; ++i) z8[i] = 0;
  __builtin_amdgcn_tensor_load_to_lds(g0, g1, z4, z4, z8, 0);
}

// ---------------------------------------------------------------------------
// Kernel 1: 3x3 SAME conv on each 16x16x3 patch -> q (=k=v), bf16, layout
// q[(b*8+h)*1024*96 + n*96 + d] where channel ch=(y*16+x)*3+co, h=ch/96, d=ch%96
// ---------------------------------------------------------------------------
__global__ __launch_bounds__(256) void conv_build_q(
    const float* __restrict__ enc, const float* __restrict__ w,
    ushort_t* __restrict__ q) {
  __shared__ float img[768];
  __shared__ float wl[81];
  const int t = blockIdx.x;  // token id 0..8191
  const int b = t >> 10;
  const int n = t & 1023;
  const int tid = threadIdx.x;
  const float* src = enc + (size_t)t * 768;
  img[tid * 3 + 0] = src[tid * 3 + 0];
  img[tid * 3 + 1] = src[tid * 3 + 1];
  img[tid * 3 + 2] = src[tid * 3 + 2];
  if (tid < 81) wl[tid] = w[tid];
  __syncthreads();
  const int y = tid >> 4, x = tid & 15;
  float o0 = 0.f, o1 = 0.f, o2 = 0.f;
#pragma unroll
  for (int ky = 0; ky < 3; ++ky) {
    int yy = y + ky - 1;
    if (yy < 0 || yy > 15) continue;
#pragma unroll
    for (int kx = 0; kx < 3; ++kx) {
      int xx = x + kx - 1;
      if (xx < 0 || xx > 15) continue;
      const float* ip = &img[(yy * 16 + xx) * 3];
      const float* wp = &wl[(ky * 3 + kx) * 9];
#pragma unroll
      for (int ci = 0; ci < 3; ++ci) {
        float iv = ip[ci];
        o0 += iv * wp[ci * 3 + 0];
        o1 += iv * wp[ci * 3 + 1];
        o2 += iv * wp[ci * 3 + 2];
      }
    }
  }
  float o[3] = {o0, o1, o2};
#pragma unroll
  for (int co = 0; co < 3; ++co) {
    int ch = tid * 3 + co;
    int h = ch / 96;
    int d = ch - h * 96;
    q[(((size_t)b * 8 + h) * 1024 + n) * 96 + d] = f2bf(o[co]);
  }
}

// ---------------------------------------------------------------------------
__global__ __launch_bounds__(256) void cvt_bf16(
    const float* __restrict__ in, ushort_t* __restrict__ out, int n) {
  int i = blockIdx.x * 256 + threadIdx.x;
  if (i < n) out[i] = f2bf(in[i]);
}

// ---------------------------------------------------------------------------
// Generic bf16 WMMA GEMM, double-buffered LDS, TDM-staged A (and B when BT).
// C[M,N] = epilogue(A[M,K] @ B).  BT=true: B source is row-major [N,K].
// Block tile 128x128, 8 waves (4 along M x 2 along N), wave tile 32x64.
// ---------------------------------------------------------------------------
enum { EPI_SF32 = 0, EPI_BF16 = 1, EPI_BIAS_RES_F32 = 2, EPI_BIAS_GELU_BF16 = 3 };

template <int EPI, bool BT, bool TDM>
__global__ __launch_bounds__(256) void gemm_bf16(
    const ushort_t* __restrict__ A, size_t aZ, int lda,
    const ushort_t* __restrict__ B, size_t bZ, int ldb,
    char* __restrict__ Cc, int zdiv, size_t cZ1, size_t cZ2, int ldc,
    int M, int N, int K,
    const float* __restrict__ bias, const float* __restrict__ resid,
    float alpha) {
  __shared__ ushort_t Al[2][128][40];  // 80B pitch: 16B aligned, bank-spread
  __shared__ ushort_t Bl[2][128][40];  // [n][k]: K contiguous per column
  const int z = blockIdx.z;
  A += (size_t)z * aZ;
  B += (size_t)z * bZ;
  Cc += (size_t)(z / zdiv) * cZ1 + (size_t)(z % zdiv) * cZ2;
  const int tid = threadIdx.x;
  const int lane = tid & 31;
  const int w = tid >> 5;
  const int wm = (w & 3) * 32;
  const int wn = (w >> 2) * 64;
  const int m0 = blockIdx.y * 128;
  const int n0 = blockIdx.x * 128;
  const int nk = K >> 5;

  auto stageA = [&](int kt, int buf) {
    if (TDM) {
      if (w == 0)
        tdm_load_2d((unsigned)(size_t)&Al[buf][0][0],
                    A + (size_t)m0 * lda + (kt << 5), lda);
    } else {
#pragma unroll
      for (int it = 0; it < 2; ++it) {
        int idx = it * 256 + tid;
        int r = idx >> 2;
        int c = (idx & 3) << 3;
        *(u32x4*)&Al[buf][r][c] =
            *(const u32x4*)(A + (size_t)(m0 + r) * lda + ((kt << 5) + c));
      }
    }
  };
  auto stageB = [&](int kt, int buf) {
    if (BT) {
      if (TDM) {
        if (w == 0)
          tdm_load_2d((unsigned)(size_t)&Bl[buf][0][0],
                      B + (size_t)n0 * ldb + (kt << 5), ldb);
      } else {
#pragma unroll
        for (int it = 0; it < 2; ++it) {
          int idx = it * 256 + tid;
          int r = idx >> 2;
          int c = (idx & 3) << 3;
          *(u32x4*)&Bl[buf][r][c] =
              *(const u32x4*)(B + (size_t)(n0 + r) * ldb + ((kt << 5) + c));
        }
      }
    } else {
      // transpose-stage: 4 bf16 along N per thread, scatter into [n][k]
#pragma unroll
      for (int it = 0; it < 4; ++it) {
        int idx = it * 256 + tid;
        int k = idx >> 5;
        int nn = (idx & 31) << 2;
        const ushort_t* gp = B + (size_t)((kt << 5) + k) * ldb + (n0 + nn);
        if (n0 + nn + 3 < N) {
          u32x2 v = *(const u32x2*)gp;
          Bl[buf][nn + 0][k] = (ushort_t)(v[0] & 0xffffu);
          Bl[buf][nn + 1][k] = (ushort_t)(v[0] >> 16);
          Bl[buf][nn + 2][k] = (ushort_t)(v[1] & 0xffffu);
          Bl[buf][nn + 3][k] = (ushort_t)(v[1] >> 16);
        } else {
#pragma unroll
          for (int e = 0; e < 4; ++e)
            Bl[buf][nn + e][k] = (n0 + nn + e < N) ? gp[e] : (ushort_t)0;
        }
      }
    }
  };

  f32x8 acc[2][4];
#pragma unroll
  for (int mi = 0; mi < 2; ++mi)
#pragma unroll
    for (int ni = 0; ni < 4; ++ni)
#pragma unroll
      for (int j = 0; j < 8; ++j) acc[mi][ni][j] = 0.f;

  stageA(0, 0);
  stageB(0, 0);
  if (TDM) { if (w == 0) __builtin_amdgcn_s_wait_tensorcnt((short)0); }
  __syncthreads();

  for (int kt = 0; kt < nk; ++kt) {
    const int cur = kt & 1;
    const int nxt = cur ^ 1;
    if (kt + 1 < nk) { stageA(kt + 1, nxt); stageB(kt + 1, nxt); }

    union Frag { bf16x16 f; u32x4 q[2]; } a[2], bfr[4];
    {
      int kba = (lane >> 4) * 8;   // lanes 0-15: K 0..7/16..23; 16-31: +8
#pragma unroll
      for (int mi = 0; mi < 2; ++mi) {
        int r = wm + mi * 16 + (lane & 15);
        a[mi].q[0] = *(const u32x4*)&Al[cur][r][kba];
        a[mi].q[1] = *(const u32x4*)&Al[cur][r][kba + 16];
      }
      int kbb = (lane >> 4) * 8;   // lanes 0-15: K 0..7/16..23; 16-31: +8
#pragma unroll
      for (int ni = 0; ni < 4; ++ni) {
        int nn = wn + ni * 16 + (lane & 15);
        bfr[ni].q[0] = *(const u32x4*)&Bl[cur][nn][kbb];
        bfr[ni].q[1] = *(const u32x4*)&Bl[cur][nn][kbb + 16];
      }
    }
#pragma unroll
    for (int mi = 0; mi < 2; ++mi)
#pragma unroll
      for (int ni = 0; ni < 4; ++ni)
        acc[mi][ni] = __builtin_amdgcn_wmma_f32_16x16x32_bf16(
            false, a[mi].f, false, bfr[ni].f, (short)0, acc[mi][ni], false, false);

    if (TDM && (kt + 1 < nk)) {
      if (w == 0) __builtin_amdgcn_s_wait_tensorcnt((short)0);
    }
    __syncthreads();
  }

  // epilogue (C frag: lanes 0-15 col=lane row=j; lanes 16-31 row=j+8)
#pragma unroll
  for (int mi = 0; mi < 2; ++mi) {
#pragma unroll
    for (int ni = 0; ni < 4; ++ni) {
      int col = n0 + wn + ni * 16 + (lane & 15);
      if (col >= N) continue;
#pragma unroll
      for (int j = 0; j < 8; ++j) {
        int row = m0 + wm + mi * 16 + ((lane >> 4) << 3) + j;
        float v = acc[mi][ni][j];
        size_t off = (size_t)row * ldc + col;
        if (EPI == EPI_SF32) {
          ((float*)Cc)[off] = v * alpha;
        } else if (EPI == EPI_BF16) {
          ((ushort_t*)Cc)[off] = f2bf(v);
        } else if (EPI == EPI_BIAS_RES_F32) {
          ((float*)Cc)[off] = v + bias[col] + resid[off];
        } else {  // bias + exact GELU -> bf16
          float t = v + bias[col];
          t = 0.5f * t * (1.0f + erff(t * 0.70710678118654752f));
          ((ushort_t*)Cc)[off] = f2bf(t);
        }
      }
    }
  }
}

// ---------------------------------------------------------------------------
// per-row softmax stats (max, 1/sumexp) over 1024 cols; one wave per row
// ---------------------------------------------------------------------------
__global__ __launch_bounds__(256) void row_stats(
    const float* __restrict__ S, float* __restrict__ stats) {
  int row = blockIdx.x * 8 + (threadIdx.x >> 5);
  int lane = threadIdx.x & 31;
  const float* p = S + (size_t)row * 1024;
  float v[32];
  float mx = -3.0e38f;
#pragma unroll
  for (int i = 0; i < 32; ++i) { v[i] = p[i * 32 + lane]; mx = fmaxf(mx, v[i]); }
  mx = wave_max(mx);
  float s = 0.f;
#pragma unroll
  for (int i = 0; i < 32; ++i) s += __expf(v[i] - mx);
  s = wave_sum(s);
  if (lane == 0) { stats[2 * row] = mx; stats[2 * row + 1] = 1.0f / s; }
}

// ---------------------------------------------------------------------------
// normalize + re-attention head mix + BN (folded), write bf16 P'
// out[g] = sum_h (bn_s[g]*W[g,h]) * P_h + (bn_s[g]*b[g] + beta[g])
// ---------------------------------------------------------------------------
__global__ __launch_bounds__(256) void mix_heads(
    const float* __restrict__ S, const float* __restrict__ stats,
    const float* __restrict__ W, const float* __restrict__ rb,
    const float* __restrict__ gamma, const float* __restrict__ beta,
    ushort_t* __restrict__ Ap) {
  __shared__ float wp[64];
  __shared__ float cg[8];
  const int tid = threadIdx.x;
  const float bns = rsqrtf(1.0f + 1.0e-3f);
  if (tid < 64) wp[tid] = gamma[tid >> 3] * bns * W[tid];
  if (tid < 8) cg[tid] = gamma[tid] * bns * rb[tid] + beta[tid];
  __syncthreads();
  size_t id = (size_t)blockIdx.x * 256 + tid;
  int m4 = (int)(id & 255) * 4;
  int n = (int)((id >> 8) & 1023);
  int b = (int)(id >> 18);
  const size_t HS = (size_t)1024 * 1024;
  size_t base = ((size_t)b * 8 * 1024 + (size_t)n) * 1024 + (size_t)m4;
  float out[8][4];
#pragma unroll
  for (int g = 0; g < 8; ++g) {
    out[g][0] = cg[g]; out[g][1] = cg[g]; out[g][2] = cg[g]; out[g][3] = cg[g];
  }
#pragma unroll
  for (int h = 0; h < 8; ++h) {
    f32x4 sv = *(const f32x4*)(S + base + (size_t)h * HS);
    size_t sr = 2 * ((size_t)(b * 8 + h) * 1024 + n);
    float mx = stats[sr];
    float rs = stats[sr + 1];
    float p0 = __expf(sv[0] - mx) * rs;
    float p1 = __expf(sv[1] - mx) * rs;
    float p2 = __expf(sv[2] - mx) * rs;
    float p3 = __expf(sv[3] - mx) * rs;
#pragma unroll
    for (int g = 0; g < 8; ++g) {
      float wg = wp[g * 8 + h];
      out[g][0] += wg * p0; out[g][1] += wg * p1;
      out[g][2] += wg * p2; out[g][3] += wg * p3;
    }
  }
#pragma unroll
  for (int g = 0; g < 8; ++g) {
    u16x4 ov;
    ov[0] = f2bf(out[g][0]); ov[1] = f2bf(out[g][1]);
    ov[2] = f2bf(out[g][2]); ov[3] = f2bf(out[g][3]);
    *(u16x4*)(Ap + base + (size_t)g * HS) = ov;
  }
}

// ---------------------------------------------------------------------------
// row LayerNorm over 768 (one wave per row); optional bf16 mirror
// ---------------------------------------------------------------------------
template <bool WB>
__global__ __launch_bounds__(256) void layernorm_row(
    const float* __restrict__ in, const float* __restrict__ g,
    const float* __restrict__ bta, float* __restrict__ outF,
    ushort_t* __restrict__ outB) {
  int row = blockIdx.x * 8 + (threadIdx.x >> 5);
  int lane = threadIdx.x & 31;
  const float* p = in + (size_t)row * 768;
  float v[24];
  float s = 0.f;
#pragma unroll
  for (int i = 0; i < 24; ++i) { v[i] = p[i * 32 + lane]; s += v[i]; }
  s = wave_sum(s);
  float mu = s * (1.0f / 768.0f);
  float q = 0.f;
#pragma unroll
  for (int i = 0; i < 24; ++i) { float d = v[i] - mu; q += d * d; }
  q = wave_sum(q);
  float rs = rsqrtf(q * (1.0f / 768.0f) + 1.0e-3f);
#pragma unroll
  for (int i = 0; i < 24; ++i) {
    int col = i * 32 + lane;
    float y = (v[i] - mu) * rs * g[col] + bta[col];
    outF[(size_t)row * 768 + col] = y;
    if (WB) outB[(size_t)row * 768 + col] = f2bf(y);
  }
}

// ---------------------------------------------------------------------------
extern "C" void kernel_launch(void* const* d_in, const int* in_sizes, int n_in,
                              void* d_out, int out_size, void* d_ws, size_t ws_size,
                              hipStream_t stream) {
  const float* enc  = (const float*)d_in[0];
  const float* qcw  = (const float*)d_in[1];
  const float* rw   = (const float*)d_in[2];
  const float* rbi  = (const float*)d_in[3];
  const float* bng  = (const float*)d_in[4];
  const float* bnb  = (const float*)d_in[5];
  const float* pw   = (const float*)d_in[6];
  const float* pb   = (const float*)d_in[7];
  const float* w1   = (const float*)d_in[8];
  const float* b1   = (const float*)d_in[9];
  const float* w2   = (const float*)d_in[10];
  const float* b2   = (const float*)d_in[11];
  const float* ln1g = (const float*)d_in[12];
  const float* ln1b = (const float*)d_in[13];
  const float* ln2g = (const float*)d_in[14];
  const float* ln2b = (const float*)d_in[15];

  char* ws = (char*)d_ws;
  size_t off = 0;
  auto alloc = [&](size_t bytes) -> char* {
    char* p = ws + off;
    off += (bytes + 255) & ~(size_t)255;
    return p;
  };
  const size_t BH = 64;  // B*H
  ushort_t* qb   = (ushort_t*)alloc(BH * 1024 * 96 * 2);      // q=k=v bf16
  float*    S    = (float*)   alloc(BH * 1024 * 1024 * 4);    // scaled scores
  float*    st   = (float*)   alloc(BH * 1024 * 2 * 4);       // (max, 1/sum)
  ushort_t* Ap   = (ushort_t*)alloc(BH * 1024 * 1024 * 2);    // mixed P' bf16
  ushort_t* X    = (ushort_t*)alloc((size_t)8192 * 768 * 2);  // attn out bf16
  ushort_t* Wp   = (ushort_t*)alloc((size_t)768 * 768 * 2);
  ushort_t* W1b  = (ushort_t*)alloc((size_t)768 * 3072 * 2);
  ushort_t* W2b  = (ushort_t*)alloc((size_t)3072 * 768 * 2);
  float*    T1   = (float*)   alloc((size_t)8192 * 768 * 4);  // proj+res
  float*    XLNf = (float*)   alloc((size_t)8192 * 768 * 4);  // LN1 out f32
  ushort_t* XLNb = (ushort_t*)alloc((size_t)8192 * 768 * 2);  // LN1 out bf16
  ushort_t* Hg   = (ushort_t*)alloc((size_t)8192 * 3072 * 2); // GELU(ffn1) bf16
  float*    T2   = (float*)   alloc((size_t)8192 * 768 * 4);  // ffn2+res
  (void)ws_size; (void)n_in; (void)in_sizes; (void)out_size;

  // weights -> bf16
  cvt_bf16<<<(768 * 768 + 255) / 256, 256, 0, stream>>>(pw, Wp, 768 * 768);
  cvt_bf16<<<(768 * 3072 + 255) / 256, 256, 0, stream>>>(w1, W1b, 768 * 3072);
  cvt_bf16<<<(3072 * 768 + 255) / 256, 256, 0, stream>>>(w2, W2b, 3072 * 768);

  // q = k = v from 3x3 conv
  conv_build_q<<<8192, 256, 0, stream>>>(enc, qcw, qb);

  // S = scale * q q^T   per (b,h);  z indexes b*8+h
  const float scale = 0.10206207261596576f;  // 96^-0.5
  gemm_bf16<EPI_SF32, true, true><<<dim3(8, 8, 64), 256, 0, stream>>>(
      qb, (size_t)1024 * 96, 96, qb, (size_t)1024 * 96, 96,
      (char*)S, 1, (size_t)1024 * 1024 * 4, 0, 1024,
      1024, 1024, 96, nullptr, nullptr, scale);

  // softmax stats + head-mix (BN folded) -> bf16 P'
  row_stats<<<8192, 256, 0, stream>>>(S, st);
  mix_heads<<<8192, 256, 0, stream>>>(S, st, rw, rbi, bng, bnb, Ap);

  // X[b,n, g*96+d] = P'[b,g] @ v[b,g]
  gemm_bf16<EPI_BF16, false, true><<<dim3(1, 8, 64), 256, 0, stream>>>(
      Ap, (size_t)1024 * 1024, 1024, qb, (size_t)1024 * 96, 96,
      (char*)X, 8, (size_t)1024 * 768 * 2, (size_t)96 * 2, 768,
      1024, 96, 1024, nullptr, nullptr, 1.0f);

  // proj + bias + residual(enc)
  gemm_bf16<EPI_BIAS_RES_F32, false, true><<<dim3(6, 64, 1), 256, 0, stream>>>(
      X, 0, 768, Wp, 0, 768,
      (char*)T1, 1, 0, 0, 768,
      8192, 768, 768, pb, enc, 1.0f);

  layernorm_row<true><<<1024, 256, 0, stream>>>(T1, ln1g, ln1b, XLNf, XLNb);

  // FFN1 + bias + exact GELU
  gemm_bf16<EPI_BIAS_GELU_BF16, false, true><<<dim3(24, 64, 1), 256, 0, stream>>>(
      XLNb, 0, 768, W1b, 0, 3072,
      (char*)Hg, 1, 0, 0, 3072,
      8192, 3072, 768, b1, nullptr, 1.0f);

  // FFN2 + bias + residual(LN1 out)
  gemm_bf16<EPI_BIAS_RES_F32, false, true><<<dim3(6, 64, 1), 256, 0, stream>>>(
      Hg, 0, 3072, W2b, 0, 768,
      (char*)T2, 1, 0, 0, 768,
      8192, 768, 3072, b2, XLNf, 1.0f);

  layernorm_row<false><<<1024, 256, 0, stream>>>(T2, ln2g, ln2b, (float*)d_out, nullptr);
}
